// MultiLayerRNN_80169859547173
// MI455X (gfx1250) — matrix operand order repaired and checked by reading, not compiled, not run on hardware
//
#include <hip/hip_runtime.h>
#include <hip/hip_bf16.h>

// MultiLayerRNN on MI455X (gfx1250, wave32, WMMA).
//
// Pipeline:
//   1) convert x + all weights fp32 -> bf16 (one pass)
//   2) GEMM (WMMA bf16 + async-LDS A-staging): xW = x @ W_ih0^T + b_ih0 + b_hh0
//   3) scan layer0: h_t = tanh(xW_t + h_{t-1} @ W_hh0^T)   (writes seq bf16)
//   4) GEMM: xW = seq @ W_ih1^T + b_ih1 + b_hh1            (buffer reused)
//   5) scan layer1: only final h kept (bf16)
//   6) GEMM: out = hT @ W_fc^T + b_fc                      (fp32 -> d_out)
//
// GEMM: A tile (64x32 bf16 = 4KB) staged to double-buffered LDS with
// global_load_async_to_lds_b128 (ASYNCcnt), overlapping next tile's DMA with
// current tile's WMMAs; waves read A fragments via ds_load_b128.
// Scan: W_hh resident in VGPRs (16 B-fragments x 8 VGPRs per wave), h passed
// between steps through a 32KB ping-pong LDS buffer, one barrier per step,
// and the 16-deep WMMA accumulation split into two independent chains to
// halve serial XDL latency on the sequential critical path.

typedef unsigned short u16;
typedef __attribute__((ext_vector_type(16))) __bf16 bf16x16;
typedef __attribute__((ext_vector_type(8)))  __bf16 bf16x8;
typedef __attribute__((ext_vector_type(8)))  float  f32x8;

__device__ __forceinline__ u16 f32_to_bf16_rne(float f) {
  unsigned u = __builtin_bit_cast(unsigned, f);
  unsigned r = u + 0x7FFFu + ((u >> 16) & 1u);
  return (u16)(r >> 16);
}

__device__ __forceinline__ bf16x16 cat16(bf16x8 lo, bf16x8 hi) {
  return __builtin_shufflevector(lo, hi, 0, 1, 2, 3, 4, 5, 6, 7,
                                         8, 9, 10, 11, 12, 13, 14, 15);
}

// Async DMA: 16 bytes global -> LDS, tracked by ASYNCcnt (CDNA5).
// Flat addresses of __shared__ objects carry the LDS byte offset in the low
// 32 bits (aperture in the high 32), so truncation yields the DS address.
__device__ __forceinline__ void async_copy_b128(const void* gsrc, void* lds_dst) {
  unsigned dst = (unsigned)(size_t)lds_dst;
  unsigned long long src = (unsigned long long)(size_t)gsrc;
  asm volatile("global_load_async_to_lds_b128 %0, %1, off"
               :: "v"(dst), "v"(src) : "memory");
}
__device__ __forceinline__ void wait_asynccnt_le1() {
  asm volatile("s_wait_asynccnt 0x1" ::: "memory");
}
__device__ __forceinline__ void wait_asynccnt_0() {
  asm volatile("s_wait_asynccnt 0x0" ::: "memory");
}

// ---------------------------------------------------------------- convert
__global__ void cvt_f32_bf16(const float* __restrict__ src,
                             u16* __restrict__ dst, int n) {
  int i = blockIdx.x * blockDim.x + threadIdx.x;
  int stride = gridDim.x * blockDim.x;
  for (; i < n; i += stride) dst[i] = f32_to_bf16_rne(src[i]);
}

// ---------------------------------------------------------------- GEMM
// C[M,N] (fp32) = A[M,K](bf16,row-major) @ W[N,K](bf16,row-major)^T + bias0 + bias1
// Workgroup: 256 threads (8 waves). Tile: 64(M) x 128(N); wave w owns N-tile w.
// M % 64 == 0, N % 128 == 0, K % 32 == 0.
__global__ __launch_bounds__(256) void wmma_gemm_bf16(
    const u16* __restrict__ A, const u16* __restrict__ W,
    const float* __restrict__ bias0, const float* __restrict__ bias1,
    float* __restrict__ C, int M, int N, int K) {
  const int tid  = (int)threadIdx.x;
  const int wave = tid >> 5;
  const int lane = tid & 31;
  const int l16  = lane & 15;
  const int lg   = lane >> 4;
  const int n0   = blockIdx.x * 128 + wave * 16;
  const int m0   = blockIdx.y * 64;

  __shared__ u16 abuf[2][64 * 32];  // double-buffered 64x32 bf16 A tile (8KB)

  // Cooperative async stage of A tile for K range [k0, k0+32).
  const int srow = tid >> 2;                 // 0..63
  const int schk = (tid & 3) * 8;            // 0,8,16,24 (bf16 elements)
  auto stage = [&](int buf, int k0) {
    async_copy_b128(A + (size_t)(m0 + srow) * K + k0 + schk,
                    &abuf[buf][srow * 32 + schk]);
  };

  f32x8 acc[4] = {f32x8{}, f32x8{}, f32x8{}, f32x8{}};

  const int nk = K >> 5;
  stage(0, 0);
  for (int i = 0; i < nk; ++i) {
    const int k0 = i << 5;
    if (i + 1 < nk) { stage((i + 1) & 1, k0 + 32); wait_asynccnt_le1(); }
    else           { wait_asynccnt_0(); }
    __syncthreads();  // A tile i fully in LDS for all waves

    // B fragment (32x16, K x N): lane holds W row (n0+l16), K = k0+16*lg .. +15
    const u16* wp = W + (size_t)(n0 + l16) * K + k0 + 16 * lg;
    bf16x16 bfrag = *reinterpret_cast<const bf16x16*>(wp);

    const u16* at = &abuf[i & 1][0];
#pragma unroll
    for (int mi = 0; mi < 4; ++mi) {
      // A fragment (16x32): row (mi*16+l16), K = 16*lg..+15
      const u16* ap = at + (mi * 16 + l16) * 32 + 16 * lg;
      bf16x8 alo = *reinterpret_cast<const bf16x8*>(ap);
      bf16x8 ahi = *reinterpret_cast<const bf16x8*>(ap + 8);
      bf16x16 afrag = cat16(alo, ahi);
      acc[mi] = __builtin_amdgcn_wmma_f32_16x16x32_bf16(
          false, afrag, false, bfrag, (short)0, acc[mi], false, false);
    }
    __syncthreads();  // reads done before this buffer is overwritten at i+1
  }

  const int n = n0 + l16;
  float bv = (bias0 ? bias0[n] : 0.f) + (bias1 ? bias1[n] : 0.f);
#pragma unroll
  for (int mi = 0; mi < 4; ++mi) {
#pragma unroll
    for (int r = 0; r < 8; ++r) {
      int row = m0 + mi * 16 + r + 8 * lg;   // C layout: M = r + 8*(lane/16)
      C[(size_t)row * N + n] = acc[mi][r] + bv;
    }
  }
}

// ---------------------------------------------------------------- scan
// One WG = 16 batch rows for all T timesteps. 1024 threads = 32 waves,
// wave w owns hidden columns [16w, 16w+16). W_hh B-fragments pinned in VGPRs.
__global__ __launch_bounds__(1024) void rnn_scan_wmma(
    const float* __restrict__ xW,   // [B, T, 512] fp32 (biases prefolded)
    const u16* __restrict__ Whh,    // [512, 512] bf16 row-major
    u16* __restrict__ seq_out,      // [B, T, 512] bf16 or nullptr
    u16* __restrict__ hT_out,       // [B, 512] bf16 or nullptr
    int T) {
  constexpr int H = 512;
  const int tid  = (int)threadIdx.x;
  const int wave = tid >> 5;          // n-tile index 0..31
  const int lane = tid & 31;
  const int l16  = lane & 15;
  const int lg   = lane >> 4;
  const int b0   = blockIdx.x * 16;   // batch slice

  __shared__ u16 hbuf[2][16][H];      // 32 KB ping-pong hidden state

  // Load W_hh B-fragments once: B[k][n] = Whh[n][k] -> contiguous 32B reads.
  bf16x16 Bf[16];
#pragma unroll
  for (int kt = 0; kt < 16; ++kt) {
    const u16* wp = Whh + (size_t)(wave * 16 + l16) * H + kt * 32 + 16 * lg;
    Bf[kt] = *reinterpret_cast<const bf16x16*>(wp);
  }

  // h_0 = 0
  {
    u16* z = &hbuf[0][0][0];
#pragma unroll
    for (int i = 0; i < 8; ++i) z[tid * 8 + i] = 0;
  }
  __syncthreads();

  const int ncol = wave * 16 + l16;
  for (int t = 0; t < T; ++t) {
    // acc0 = xW[:, t, ncol] in C layout (M = r + 8*lg, N = l16)
    f32x8 acc0, acc1 = f32x8{};
#pragma unroll
    for (int r = 0; r < 8; ++r)
      acc0[r] = xW[(size_t)(b0 + r + 8 * lg) * T * H + (size_t)t * H + ncol];

    if (t + 1 < T)
      __builtin_prefetch(
          &xW[(size_t)(b0 + 8 * lg) * T * H + (size_t)(t + 1) * H + ncol], 0, 1);

    // acc += h_{t-1} @ W_hh^T, as two independent 8-deep WMMA chains to
    // halve the serial XDL latency on this sequential critical path.
    const u16* hrow = &hbuf[t & 1][l16][0];
#pragma unroll
    for (int kt = 0; kt < 8; ++kt) {
      const u16* hp = hrow + kt * 32 + 16 * lg;
      bf16x8 alo = *reinterpret_cast<const bf16x8*>(hp);
      bf16x8 ahi = *reinterpret_cast<const bf16x8*>(hp + 8);
      acc0 = __builtin_amdgcn_wmma_f32_16x16x32_bf16(
          false, cat16(alo, ahi), false, Bf[kt], (short)0, acc0, false, false);
      const u16* hq = hrow + (kt + 8) * 32 + 16 * lg;
      bf16x8 blo = *reinterpret_cast<const bf16x8*>(hq);
      bf16x8 bhi = *reinterpret_cast<const bf16x8*>(hq + 8);
      acc1 = __builtin_amdgcn_wmma_f32_16x16x32_bf16(
          false, cat16(blo, bhi), false, Bf[kt + 8], (short)0, acc1, false, false);
    }

    // h_t = tanh(acc0+acc1); publish to other buffer + (optionally) to memory
#pragma unroll
    for (int r = 0; r < 8; ++r) {
      float v = tanhf(acc0[r] + acc1[r]);
      u16 bv = f32_to_bf16_rne(v);
      int m = r + 8 * lg;
      hbuf[(t + 1) & 1][m][ncol] = bv;
      if (seq_out)
        seq_out[(size_t)(b0 + m) * T * H + (size_t)t * H + ncol] = bv;
      if (hT_out && t == T - 1)
        hT_out[(size_t)(b0 + m) * H + ncol] = bv;
    }
    __syncthreads();  // writes to buf[(t+1)&1] visible before t+1 reads
  }
}

// ---------------------------------------------------------------- launch
extern "C" void kernel_launch(void* const* d_in, const int* in_sizes, int n_in,
                              void* d_out, int out_size, void* d_ws,
                              size_t ws_size, hipStream_t stream) {
  (void)in_sizes; (void)n_in; (void)out_size; (void)ws_size;
  constexpr int Bb = 64, Tt = 1024, Dd = 256, Hh = 512, Oo = 256;

  const float* x     = (const float*)d_in[0];
  const float* W_ih0 = (const float*)d_in[1];
  const float* W_hh0 = (const float*)d_in[2];
  const float* b_ih0 = (const float*)d_in[3];
  const float* b_hh0 = (const float*)d_in[4];
  const float* W_ih1 = (const float*)d_in[5];
  const float* W_hh1 = (const float*)d_in[6];
  const float* b_ih1 = (const float*)d_in[7];
  const float* b_hh1 = (const float*)d_in[8];
  const float* W_fc  = (const float*)d_in[9];
  const float* b_fc  = (const float*)d_in[10];

  // Workspace layout (~226 MB total)
  char* w = (char*)d_ws;
  size_t off = 0;
  auto take = [&](size_t bytes) -> void* {
    void* p = w + off;
    off = (off + bytes + 255) & ~(size_t)255;
    return p;
  };
  u16*   x_bf  = (u16*)take((size_t)Bb * Tt * Dd * 2);
  u16*   wih0b = (u16*)take((size_t)Hh * Dd * 2);
  u16*   whh0b = (u16*)take((size_t)Hh * Hh * 2);
  u16*   wih1b = (u16*)take((size_t)Hh * Hh * 2);
  u16*   whh1b = (u16*)take((size_t)Hh * Hh * 2);
  u16*   wfcb  = (u16*)take((size_t)Oo * Hh * 2);
  float* xw    = (float*)take((size_t)Bb * Tt * Hh * 4);  // reused both layers
  u16*   seq   = (u16*)take((size_t)Bb * Tt * Hh * 2);    // layer0 outputs
  u16*   ht    = (u16*)take((size_t)Bb * Hh * 2);         // final hidden

  auto cvt = [&](const float* s, u16* d, int n) {
    int blocks = (n + 255) / 256;
    if (blocks > 4096) blocks = 4096;
    cvt_f32_bf16<<<blocks, 256, 0, stream>>>(s, d, n);
  };
  cvt(x,     x_bf,  Bb * Tt * Dd);
  cvt(W_ih0, wih0b, Hh * Dd);
  cvt(W_hh0, whh0b, Hh * Hh);
  cvt(W_ih1, wih1b, Hh * Hh);
  cvt(W_hh1, whh1b, Hh * Hh);
  cvt(W_fc,  wfcb,  Oo * Hh);

  // Layer 0: input projection + scan
  wmma_gemm_bf16<<<dim3(Hh / 128, (Bb * Tt) / 64), 256, 0, stream>>>(
      x_bf, wih0b, b_ih0, b_hh0, xw, Bb * Tt, Hh, Dd);
  rnn_scan_wmma<<<Bb / 16, 1024, 0, stream>>>(xw, whh0b, seq, nullptr, Tt);

  // Layer 1: input projection + scan (only final hidden kept)
  wmma_gemm_bf16<<<dim3(Hh / 128, (Bb * Tt) / 64), 256, 0, stream>>>(
      seq, wih1b, b_ih1, b_hh1, xw, Bb * Tt, Hh, Hh);
  rnn_scan_wmma<<<Bb / 16, 1024, 0, stream>>>(xw, whh1b, nullptr, ht, Tt);

  // Final FC: [64,512] @ [256,512]^T + b_fc -> d_out fp32
  wmma_gemm_bf16<<<dim3(Oo / 128, Bb / 64), 256, 0, stream>>>(
      ht, wfcb, b_fc, nullptr, (float*)d_out, Bb, Oo, Hh);
}
